// Net_80685255622993
// MI455X (gfx1250) — matrix operand order repaired and checked by reading, not compiled, not run on hardware
//
#include <hip/hip_runtime.h>

typedef __bf16 bf16;
typedef __attribute__((ext_vector_type(16))) __bf16 v16bf;
typedef __attribute__((ext_vector_type(8)))  __bf16 v8bf;
typedef __attribute__((ext_vector_type(8)))  float  v8f;

#define BATCH 32
#define NPTS  8192
#define G40   40

// ---------------------------------------------------------------------------
// Weight repack: f32 OIDHW -> bf16 per-lane WMMA B-fragments
// layout: [ntile][kchunk][lane(32)][16 bf16], K = tap*CIp + ci (tap-major)
// ---------------------------------------------------------------------------
template<int CIP, int CIRAW, int CO, int KS>
__global__ __launch_bounds__(256) void pack_w_kernel(const float* __restrict__ w,
                                                     bf16* __restrict__ wp) {
    constexpr int KT = KS * KS * KS * CIP;
    constexpr int KC = KT / 32;
    constexpr int NT = CO / 16;
    long long idx = (long long)blockIdx.x * blockDim.x + threadIdx.x;
    long long total = (long long)NT * KC * 32 * 16;
    if (idx >= total) return;
    int e    = (int)(idx & 15);
    int lane = (int)((idx >> 4) & 31);
    long long t = idx >> 9;
    int kc = (int)(t % KC);
    int nt = (int)(t / KC);
    int co = nt * 16 + (lane & 15);
    int k  = kc * 32 + (lane >> 4) * 16 + e;
    int tap = k / CIP, ci = k % CIP;
    int kd = tap / (KS * KS), kh = (tap / KS) % KS, kw = tap % KS;
    float v = 0.0f;
    if (ci < CIRAW)
        v = w[((((long long)co * CIRAW + ci) * KS + kd) * KS + kh) * KS + kw];
    wp[idx] = (bf16)v;
}

// ---------------------------------------------------------------------------
// Point encode + scatter-add voxelization
// ---------------------------------------------------------------------------
__global__ __launch_bounds__(256) void encode_scatter_kernel(const float* __restrict__ x,
                                                             float* __restrict__ fsum,
                                                             float* __restrict__ cnt) {
    int p = blockIdx.x * blockDim.x + threadIdx.x;
    if (p >= BATCH * NPTS) return;
    int b = p >> 13;
    float c0 = x[(long long)p * 3 + 0];
    float c1 = x[(long long)p * 3 + 1];
    float c2 = x[(long long)p * 3 + 2];
    const float PI = 3.14159265358979323846f;
    float crd[3] = {c0, c1, c2};
    int vi[3];
    for (int i = 0; i < 3; ++i) {
        int q = (int)floorf(crd[i] / 0.05f);
        q = q < -20 ? -20 : (q > 19 ? 19 : q);
        vi[i] = q + 20;
    }
    int lin = (vi[0] * G40 + vi[1]) * G40 + vi[2];
    long long base = (long long)b * (G40 * G40 * G40) + lin;
    atomicAdd(&cnt[base], 1.0f);
    float enc[24];
    for (int i = 0; i < 3; ++i) {
        float f = PI;
        for (int j = 0; j < 4; ++j) {
            float ang = crd[i] * f;
            enc[i * 8 + j]     = sinf(ang);
            enc[i * 8 + 4 + j] = cosf(ang);
            f *= 2.0f;
        }
    }
    float* fp = fsum + base * 24;
    for (int c = 0; c < 24; ++c) atomicAdd(&fp[c], enc[c]);
}

// mean-pool finalize -> bf16 feat (padded to 32 ch) + f32 mask
__global__ __launch_bounds__(256) void finalize_vox_kernel(const float* __restrict__ fsum,
                                                           const float* __restrict__ cnt,
                                                           bf16* __restrict__ feat,
                                                           float* __restrict__ mask) {
    long long v = (long long)blockIdx.x * blockDim.x + threadIdx.x;
    long long tot = (long long)BATCH * G40 * G40 * G40;
    if (v >= tot) return;
    float c = cnt[v];
    float inv = c > 0.0f ? 1.0f / c : 0.0f;
    mask[v] = c > 0.0f ? 1.0f : 0.0f;
    bf16* fp = feat + v * 32;
    const float* sp = fsum + v * 24;
    for (int ch = 0; ch < 24; ++ch) fp[ch] = (bf16)(sp[ch] * inv);
    for (int ch = 24; ch < 32; ++ch) fp[ch] = (bf16)0.0f;
}

// 2x2x2 mask max-pool
template<int DO>
__global__ __launch_bounds__(256) void maskpool_kernel(const float* __restrict__ mi,
                                                       float* __restrict__ mo) {
    int i = blockIdx.x * blockDim.x + threadIdx.x;
    int tot = BATCH * DO * DO * DO;
    if (i >= tot) return;
    int b = i / (DO * DO * DO), r = i % (DO * DO * DO);
    int od = r / (DO * DO), oh = (r / DO) % DO, ow = r % DO;
    const int DI = DO * 2;
    float m = 0.0f;
    for (int dz = 0; dz < 2; ++dz)
        for (int dy = 0; dy < 2; ++dy)
            for (int dx = 0; dx < 2; ++dx) {
                long long idx = (((long long)b * DI + (od * 2 + dz)) * DI + (oh * 2 + dy)) * DI
                                + (ow * 2 + dx);
                float t = mi[idx];
                m = t > m ? t : m;
            }
    mo[i] = m;
}

// ---------------------------------------------------------------------------
// Implicit-GEMM conv via WMMA bf16.
// One wave = 16-voxel x 32-outch tile (two 16x16 WMMA accumulators sharing
// one A fragment). Tap-outer / channel-chunk-inner loop so per-tap address &
// validity math is hoisted and inner loads walk contiguous 64-B steps.
// EPI: 0 = f32 out, *mask            (c1,c2,c3 before LN)
//      1 = bf16 out, *mask, ReLU     (s1,s2)
//      2 = bf16 out, ReLU            (c4, f1)
//      3 = f32 out permuted [b][co*8+s], no ReLU  (s4)
// ---------------------------------------------------------------------------
template<int CI, int CO, int KS, int STR, int PAD, int DIN, int DOUT, int EPI>
__global__ __launch_bounds__(256)
void conv_wmma_kernel(const bf16* __restrict__ x, const bf16* __restrict__ wp,
                      const float* __restrict__ bias, const float* __restrict__ mask,
                      float* __restrict__ of32, bf16* __restrict__ obf) {
    constexpr int KT = KS * KS * KS * CI;
    constexpr int KC = KT / 32;        // total K chunks
    constexpr int CC = CI / 32;        // K chunks per tap
    constexpr int NP = CO / 32;        // 32-wide N pair tiles
    constexpr int OV = DOUT * DOUT * DOUT;
    const int Mtotal = BATCH * OV;
    const int Mtiles = (Mtotal + 15) / 16;

    int wave = threadIdx.x >> 5;
    int lane = threadIdx.x & 31;
    long long tile = (long long)blockIdx.x * 8 + wave;
    if (tile >= (long long)Mtiles * NP) return;
    int mt = (int)(tile / NP);
    int np = (int)(tile % NP);
    int half = lane >> 4, nl = lane & 15;

    // A row handled by this lane
    int m = mt * 16 + nl;
    bool mvalid = m < Mtotal;
    int mm = mvalid ? m : 0;
    int b  = mm / OV;
    int r  = mm % OV;
    int od = r / (DOUT * DOUT), oh = (r / DOUT) % DOUT, ow = r % DOUT;

    v8f acc0 = {}, acc1 = {};
    const bf16* wptr0 = wp + ((long long)(2 * np) * KC) * 512 + lane * 16;
    const bf16* wptr1 = wptr0 + (long long)KC * 512;

    for (int tap = 0; tap < KS * KS * KS; ++tap) {
        int kd = tap / (KS * KS), kh = (tap / KS) % KS, kw = tap % KS;
        int id = od * STR + kd - PAD;
        int ih = oh * STR + kh - PAD;
        int iw = ow * STR + kw - PAD;
        bool valid = mvalid && id >= 0 && id < DIN && ih >= 0 && ih < DIN && iw >= 0 && iw < DIN;
        const bf16* xp = x + ((((long long)b * DIN + id) * DIN + ih) * DIN + iw)
                               * (long long)CI + half * 16;
        const bf16* bp0 = wptr0 + (long long)tap * CC * 512;
        const bf16* bp1 = wptr1 + (long long)tap * CC * 512;

        for (int cc = 0; cc < CC; ++cc) {
            v16bf a = {};
            if (valid) {
                *(v8bf*)&a       = *(const v8bf*)(xp + cc * 32);
                *((v8bf*)&a + 1) = *(const v8bf*)(xp + cc * 32 + 8);
            }
            v16bf b0 = *(const v16bf*)(bp0 + (long long)cc * 512);
            v16bf b1 = *(const v16bf*)(bp1 + (long long)cc * 512);
            __builtin_prefetch(bp0 + (long long)(cc + 1) * 512, 0, 0);
            acc0 = __builtin_amdgcn_wmma_f32_16x16x32_bf16(false, a, false, b0,
                                                           (short)0, acc0, false, false);
            acc1 = __builtin_amdgcn_wmma_f32_16x16x32_bf16(false, a, false, b1,
                                                           (short)0, acc1, false, false);
        }
    }

    int co0 = np * 32 + nl;
    int co1 = co0 + 16;
    float bv0 = bias[co0];
    float bv1 = bias[co1];
#pragma unroll
    for (int v = 0; v < 8; ++v) {
        int mr = mt * 16 + v + 8 * half;
        if (mr >= Mtotal) continue;
        float v0 = acc0[v] + bv0;
        float v1 = acc1[v] + bv1;
        if constexpr (EPI == 0) {
            float mk = mask[mr];
            of32[(long long)mr * CO + co0] = v0 * mk;
            of32[(long long)mr * CO + co1] = v1 * mk;
        } else if constexpr (EPI == 1) {
            float mk = mask[mr];
            v0 *= mk; v1 *= mk;
            v0 = v0 > 0.0f ? v0 : 0.0f;
            v1 = v1 > 0.0f ? v1 : 0.0f;
            obf[(long long)mr * CO + co0] = (bf16)v0;
            obf[(long long)mr * CO + co1] = (bf16)v1;
        } else if constexpr (EPI == 2) {
            v0 = v0 > 0.0f ? v0 : 0.0f;
            v1 = v1 > 0.0f ? v1 : 0.0f;
            obf[(long long)mr * CO + co0] = (bf16)v0;
            obf[(long long)mr * CO + co1] = (bf16)v1;
        } else {
            int bb = mr / 8, s = mr % 8;
            of32[(long long)bb * 8192 + co0 * 8 + s] = v0;
            of32[(long long)bb * 8192 + co1 * 8 + s] = v1;
        }
    }
}

// ---------------------------------------------------------------------------
// Per-row LayerNorm (over CO) + affine + optional mask + ReLU, bf16 out
// one block (128 thr = 4 wave32) per row
// ---------------------------------------------------------------------------
template<int CO>
__global__ __launch_bounds__(128)
void ln_relu_kernel(const float* __restrict__ xin, const float* __restrict__ mask,
                    const float* __restrict__ g, const float* __restrict__ bb,
                    bf16* __restrict__ out) {
    long long v = blockIdx.x;
    const float* xp = xin + v * CO;
    float s = 0.0f, q = 0.0f;
    for (int c = threadIdx.x; c < CO; c += 128) {
        float t = xp[c];
        s += t; q += t * t;
    }
    for (int o = 16; o > 0; o >>= 1) {
        s += __shfl_xor(s, o, 32);
        q += __shfl_xor(q, o, 32);
    }
    __shared__ float ws1[4], ws2[4];
    int w = threadIdx.x >> 5, l = threadIdx.x & 31;
    if (l == 0) { ws1[w] = s; ws2[w] = q; }
    __syncthreads();
    float ts = ws1[0] + ws1[1] + ws1[2] + ws1[3];
    float tq = ws2[0] + ws2[1] + ws2[2] + ws2[3];
    float mu = ts / CO;
    float var = tq / CO - mu * mu;
    float rs = rsqrtf(var + 1e-5f);
    float mk = mask ? mask[v] : 1.0f;
    for (int c = threadIdx.x; c < CO; c += 128) {
        float y = (xp[c] - mu) * rs * g[c] + bb[c];
        y *= mk;
        y = y > 0.0f ? y : 0.0f;
        out[v * CO + c] = (bf16)y;
    }
}

// ---------------------------------------------------------------------------
// Final head: logits = a1 @ f2w^T + f2b ; log_softmax
// ---------------------------------------------------------------------------
__global__ __launch_bounds__(64) void head_kernel(const bf16* __restrict__ a,
                                                  const float* __restrict__ w,
                                                  const float* __restrict__ bias,
                                                  float* __restrict__ out) {
    int b = blockIdx.x;
    int o = threadIdx.x;
    __shared__ float logits[40];
    if (o < 40) {
        float s = bias[o];
        const bf16* ap = a + (long long)b * 1024;
        const float* wp = w + (long long)o * 1024;
        for (int k = 0; k < 1024; ++k) s += (float)ap[k] * wp[k];
        logits[o] = s;
    }
    __syncthreads();
    if (o < 40) {
        float mx = -1e30f;
        for (int i = 0; i < 40; ++i) mx = logits[i] > mx ? logits[i] : mx;
        float se = 0.0f;
        for (int i = 0; i < 40; ++i) se += expf(logits[i] - mx);
        out[(long long)b * 40 + o] = logits[o] - mx - logf(se);
    }
}

// ---------------------------------------------------------------------------
static inline long long cdivll(long long a, long long b) { return (a + b - 1) / b; }

extern "C" void kernel_launch(void* const* d_in, const int* in_sizes, int n_in,
                              void* d_out, int out_size, void* d_ws, size_t ws_size,
                              hipStream_t stream) {
    (void)in_sizes; (void)n_in; (void)out_size; (void)ws_size;
    const float* x   = (const float*)d_in[0];
    const float* c1w = (const float*)d_in[1];  const float* c1b = (const float*)d_in[2];
    const float* g1  = (const float*)d_in[3];  const float* b1  = (const float*)d_in[4];
    const float* s1w = (const float*)d_in[5];  const float* s1b = (const float*)d_in[6];
    const float* c2w = (const float*)d_in[7];  const float* c2b = (const float*)d_in[8];
    const float* g2  = (const float*)d_in[9];  const float* b2  = (const float*)d_in[10];
    const float* s2w = (const float*)d_in[11]; const float* s2b = (const float*)d_in[12];
    const float* c3w = (const float*)d_in[13]; const float* c3b = (const float*)d_in[14];
    const float* g3  = (const float*)d_in[15]; const float* b3  = (const float*)d_in[16];
    const float* c4w = (const float*)d_in[17]; const float* c4b = (const float*)d_in[18];
    const float* s4w = (const float*)d_in[19]; const float* s4b = (const float*)d_in[20];
    const float* g4  = (const float*)d_in[21]; const float* b4  = (const float*)d_in[22];
    const float* f1w = (const float*)d_in[23]; const float* f1b = (const float*)d_in[24];
    const float* f2w = (const float*)d_in[25]; const float* f2b = (const float*)d_in[26];
    float* out = (float*)d_out;

    // ---- workspace bump allocator (256-B aligned) ----
    size_t off = 0;
    char* base = (char*)d_ws;
    auto alloc = [&](size_t bytes) -> char* {
        off = (off + 255) & ~(size_t)255;
        char* p = base + off;
        off += bytes;
        return p;
    };
    const long long V0 = (long long)BATCH * 64000;   // 40^3 voxels * batch
    const long long V1 = (long long)BATCH * 8000;    // 20^3
    const long long V2 = (long long)BATCH * 1000;    // 10^3

    bf16* wp_c1 = (bf16*)alloc(2  * 27  * 512 * 2);
    bf16* wp_s1 = (bf16*)alloc(4  * 8   * 512 * 2);
    bf16* wp_c2 = (bf16*)alloc(8  * 54  * 512 * 2);
    bf16* wp_s2 = (bf16*)alloc(16 * 32  * 512 * 2);
    bf16* wp_c3 = (bf16*)alloc(32 * 216 * 512 * 2);
    bf16* wp_c4 = (bf16*)alloc(64 * 432 * 512 * 2);
    bf16* wp_s4 = (bf16*)alloc(64 * 256 * 512 * 2);
    bf16* wp_f1 = (bf16*)alloc(64 * 256 * 512 * 2);

    float* tmpf  = (float*)alloc((size_t)V0 * 32 * 4);    // f32 conv scratch (max = c1)
    float* fsum  = tmpf;                                  // alias: dead before c1 writes
    float* cnt   = tmpf + V0 * 24;
    bf16*  hA    = (bf16*)alloc((size_t)V0 * 32 * 2);     // feat0 / h1 / h3 / h5
    bf16*  hB    = (bf16*)alloc((size_t)V1 * 64 * 2);     // h2 / h4 / h6
    float* mask0 = (float*)alloc((size_t)V0 * 4);
    float* mask1 = (float*)alloc((size_t)V1 * 4);
    float* mask2 = (float*)alloc((size_t)V2 * 4);
    float* h7    = (float*)alloc((size_t)BATCH * 8192 * 4);
    bf16*  a0    = (bf16*)alloc((size_t)BATCH * 8192 * 2);
    bf16*  a1    = (bf16*)alloc((size_t)BATCH * 1024 * 2);

    // ---- weight repack (all independent) ----
    auto packgrid = [](long long n) { return dim3((unsigned)cdivll(n, 256)); };
    pack_w_kernel<32,   24,   32,   3><<<packgrid(2  * 27  * 512), 256, 0, stream>>>(c1w, wp_c1);
    pack_w_kernel<32,   32,   64,   2><<<packgrid(4  * 8   * 512), 256, 0, stream>>>(s1w, wp_s1);
    pack_w_kernel<64,   64,   128,  3><<<packgrid(8  * 54  * 512), 256, 0, stream>>>(c2w, wp_c2);
    pack_w_kernel<128,  128,  256,  2><<<packgrid(16 * 32  * 512), 256, 0, stream>>>(s2w, wp_s2);
    pack_w_kernel<256,  256,  512,  3><<<packgrid(32 * 216 * 512), 256, 0, stream>>>(c3w, wp_c3);
    pack_w_kernel<512,  512,  1024, 3><<<packgrid(64 * 432 * 512), 256, 0, stream>>>(c4w, wp_c4);
    pack_w_kernel<1024, 1024, 1024, 2><<<packgrid(64 * 256 * 512), 256, 0, stream>>>(s4w, wp_s4);
    pack_w_kernel<8192, 8192, 1024, 1><<<packgrid(64 * 256 * 512), 256, 0, stream>>>(f1w, wp_f1);

    // ---- voxelize ----
    hipMemsetAsync(fsum, 0, (size_t)(V0 * 24 + V0) * 4, stream);  // fsum + cnt contiguous
    encode_scatter_kernel<<<dim3((unsigned)cdivll((long long)BATCH * NPTS, 256)), 256, 0, stream>>>(
        x, fsum, cnt);
    finalize_vox_kernel<<<dim3((unsigned)cdivll(V0, 256)), 256, 0, stream>>>(fsum, cnt, hA, mask0);

    auto convgrid = [](long long Mtotal, int NP) {
        return dim3((unsigned)cdivll(cdivll(Mtotal, 16) * NP, 8));
    };

    // c1: 24(->32) -> 32, 3^3 SAME @40^3, *mask0 -> f32 tmp
    conv_wmma_kernel<32, 32, 3, 1, 1, 40, 40, 0><<<convgrid(V0, 1), 256, 0, stream>>>(
        hA, wp_c1, c1b, mask0, tmpf, nullptr);
    ln_relu_kernel<32><<<dim3((unsigned)V0), 128, 0, stream>>>(tmpf, mask0, g1, b1, hA);   // h1
    maskpool_kernel<20><<<dim3((unsigned)cdivll(V1, 256)), 256, 0, stream>>>(mask0, mask1);

    // s1: 32 -> 64, 2^3 stride2, 40->20, relu((conv)*mask1)
    conv_wmma_kernel<32, 64, 2, 2, 0, 40, 20, 1><<<convgrid(V1, 2), 256, 0, stream>>>(
        hA, wp_s1, s1b, mask1, nullptr, hB);                                               // h2

    // c2: 64 -> 128, 3^3 SAME @20^3
    conv_wmma_kernel<64, 128, 3, 1, 1, 20, 20, 0><<<convgrid(V1, 4), 256, 0, stream>>>(
        hB, wp_c2, c2b, mask1, tmpf, nullptr);
    ln_relu_kernel<128><<<dim3((unsigned)V1), 128, 0, stream>>>(tmpf, mask1, g2, b2, hA);  // h3
    maskpool_kernel<10><<<dim3((unsigned)cdivll(V2, 256)), 256, 0, stream>>>(mask1, mask2);

    // s2: 128 -> 256, 2^3 stride2, 20->10
    conv_wmma_kernel<128, 256, 2, 2, 0, 20, 10, 1><<<convgrid(V2, 8), 256, 0, stream>>>(
        hA, wp_s2, s2b, mask2, nullptr, hB);                                               // h4

    // c3: 256 -> 512, 3^3 SAME @10^3
    conv_wmma_kernel<256, 512, 3, 1, 1, 10, 10, 0><<<convgrid(V2, 16), 256, 0, stream>>>(
        hB, wp_c3, c3b, mask2, tmpf, nullptr);
    ln_relu_kernel<512><<<dim3((unsigned)V2), 128, 0, stream>>>(tmpf, mask2, g3, b3, hA);  // h5

    // c4: 512 -> 1024, 3^3 stride2 VALID, 10->4, dense, relu
    conv_wmma_kernel<512, 1024, 3, 2, 0, 10, 4, 2><<<convgrid((long long)BATCH * 64, 32), 256, 0,
                                                     stream>>>(hA, wp_c4, c4b, nullptr, nullptr, hB);

    // s4: 1024 -> 1024, 2^3 stride2, 4->2, write NCDHW-flattened f32 [B][8192]
    conv_wmma_kernel<1024, 1024, 2, 2, 0, 4, 2, 3><<<convgrid((long long)BATCH * 8, 32), 256, 0,
                                                     stream>>>(hB, wp_s4, s4b, nullptr, h7, nullptr);

    // LN over 8192 + relu -> bf16
    ln_relu_kernel<8192><<<dim3(BATCH), 128, 0, stream>>>(h7, nullptr, g4, b4, a0);

    // f1: [32,8192] x [8192,1024] + relu, as KS=1 conv
    conv_wmma_kernel<8192, 1024, 1, 1, 0, 1, 1, 2><<<convgrid(BATCH, 32), 256, 0, stream>>>(
        a0, wp_f1, f1b, nullptr, nullptr, a1);

    // f2 + log_softmax
    head_kernel<<<dim3(BATCH), 64, 0, stream>>>(a1, f2w, f2b, out);
}
